// StructureModuleLayer_36395552866968
// MI455X (gfx1250) — matrix operand order repaired and checked by reading, not compiled, not run on hardware
//
#include <hip/hip_runtime.h>
#include <hip/hip_bf16.h>
#include <math.h>

typedef __bf16 bf16_t;
typedef __attribute__((ext_vector_type(16))) __bf16 v16bf;
typedef __attribute__((ext_vector_type(8)))  float  v8f;
typedef __attribute__((ext_vector_type(4)))  float  f4;
typedef __attribute__((ext_vector_type(4)))  unsigned int u32x4;
typedef __attribute__((ext_vector_type(8)))  int i32x8;
typedef __attribute__((ext_vector_type(4)))  int i32x4;

#define N_    768
#define CS_   384
#define CZ_   128
#define H_    12
#define C_    16
#define PQ_   4
#define PV_   8
#define FEAT_ 2112
#define WC_f  0.23570226039551587f   /* sqrt(2/(9*PQ)) */
#define WL_f  0.57735026918962576f   /* sqrt(1/3)      */

static __device__ __forceinline__ bf16_t f2bf(float f) {
    unsigned u = __builtin_bit_cast(unsigned, f);
    unsigned r = u + 0x7FFFu + ((u >> 16) & 1u);
    unsigned short h = (unsigned short)(r >> 16);
    return __builtin_bit_cast(bf16_t, h);
}
static __device__ __forceinline__ float bf2f(bf16_t b) {
    unsigned short s = __builtin_bit_cast(unsigned short, b);
    unsigned u = ((unsigned)s) << 16;
    return __builtin_bit_cast(float, u);
}

static __device__ __forceinline__ v8f wmma_bf16(v16bf a, v16bf b, v8f c) {
    return __builtin_amdgcn_wmma_f32_16x16x32_bf16(false, a, false, b, (short)0, c, false, false);
}

// A fragment (16x32, bf16, row-major source with leading-dim ldk).
// lane<16: M=lane,    K chunks [k0..k0+7] and [k0+16..k0+23]
// lane>=16: M=lane-16, K chunks [k0+8..k0+15] and [k0+24..k0+31]
static __device__ __forceinline__ v16bf load_a_frag(const bf16_t* A, int ldk,
                                                    int m_base, int k0, int lane) {
    int m = m_base + (lane & 15);
    const bf16_t* p = A + (size_t)m * ldk + k0 + ((lane >> 4) << 3);
    f4 lo = *(const f4*)p;
    f4 hi = *(const f4*)(p + 16);
    v16bf r;
    ((f4*)&r)[0] = lo;
    ((f4*)&r)[1] = hi;
    return r;
}

// B fragment (32x16, bf16) from transposed storage BT[N][K] (leading-dim ldk).
// lane<16: N=lane, K chunks [k0..k0+7] and [k0+16..k0+23] ; lane>=16: N=lane-16, K chunks [k0+8..k0+15] and [k0+24..k0+31]
static __device__ __forceinline__ v16bf load_b_frag(const bf16_t* BT, int ldk,
                                                    int n_base, int k0, int lane) {
    int n = n_base + (lane & 15);
    const bf16_t* p = BT + (size_t)n * ldk + k0 + ((lane >> 4) << 3);
    f4 lo = *(const f4*)p;
    f4 hi = *(const f4*)(p + 16);
    v16bf r;
    ((f4*)&r)[0] = lo;
    ((f4*)&r)[1] = hi;
    return r;
}

// ---------------- utility kernels ----------------
__global__ void k_cvt_bf16(const float* __restrict__ in, bf16_t* __restrict__ out, int n) {
    int i = blockIdx.x * blockDim.x + threadIdx.x;
    if (i < n) out[i] = f2bf(in[i]);
}

__global__ void k_zero_bf16(bf16_t* out, int n) {
    int i = blockIdx.x * blockDim.x + threadIdx.x;
    if (i < n) out[i] = __builtin_bit_cast(bf16_t, (unsigned short)0);
}

// in: (K x O) f32 row-major -> out: (O x K) bf16 row-major
__global__ void k_transpose(const float* __restrict__ in, bf16_t* __restrict__ out,
                            int K, int O) {
    int idx = blockIdx.x * blockDim.x + threadIdx.x;
    if (idx >= K * O) return;
    int kk = idx / O, o = idx % O;
    out[(size_t)o * K + kk] = f2bf(in[idx]);
}

// ---------------- generic WMMA GEMM: C = A(MxK)bf16 @ BT(NpxK)bf16 + bias ----------------
__global__ void k_gemm(const bf16_t* __restrict__ A, const bf16_t* __restrict__ BT,
                       const float* __restrict__ bias, float* __restrict__ Cout,
                       bf16_t* __restrict__ Cbf, int M, int K, int Np, int relu) {
    int lane = threadIdx.x;
    int m0 = blockIdx.x * 16, n0 = blockIdx.y * 16;
    v8f acc = {0, 0, 0, 0, 0, 0, 0, 0};
    for (int k0 = 0; k0 < K; k0 += 32) {
        v16bf a = load_a_frag(A, K, m0, k0, lane);
        v16bf b = load_b_frag(BT, K, n0, k0, lane);
        acc = wmma_bf16(a, b, acc);
    }
    int n = n0 + (lane & 15);
    float bv = bias ? bias[n] : 0.f;
    for (int r = 0; r < 8; r++) {
        int m = m0 + r + ((lane >> 4) << 3);
        float v = acc[r] + bv;
        if (relu) v = v > 0.f ? v : 0.f;
        Cout[(size_t)m * Np + n] = v;
        if (Cbf) Cbf[(size_t)m * Np + n] = f2bf(v);
    }
}

// ---------------- point prep: global frames, qhat/khat (32-dim), vB, qn/kn ----------------
__global__ void k_prep(const float* __restrict__ q, const float* __restrict__ k,
                       const float* __restrict__ v, const float* __restrict__ qp,
                       const float* __restrict__ kp, const float* __restrict__ vp,
                       const float* __restrict__ rot, const float* __restrict__ trans,
                       const float* __restrict__ head_w,
                       bf16_t* __restrict__ qhat, bf16_t* __restrict__ khat,
                       bf16_t* __restrict__ vB, float* __restrict__ qn,
                       float* __restrict__ kn) {
    int i = blockIdx.x;
    int tid = threadIdx.x; // 192 threads
    __shared__ float R[9], t[3], sqn[H_], skn[H_], sg[H_];
    if (tid < 9) R[tid] = rot[(size_t)i * 9 + tid];
    if (tid < 3) t[tid] = trans[(size_t)i * 3 + tid];
    if (tid < H_) {
        sqn[tid] = 0.f; skn[tid] = 0.f;
        float gam = logf(1.f + expf(head_w[tid]));
        sg[tid] = sqrtf(WC_f * gam);
    }
    __syncthreads();
    if (tid < 48) { // (h,p) for q/k points
        int h = tid >> 2;
        const float* lq = qp + (size_t)i * 144 + tid * 3;
        float gx = R[0]*lq[0] + R[1]*lq[1] + R[2]*lq[2] + t[0];
        float gy = R[3]*lq[0] + R[4]*lq[1] + R[5]*lq[2] + t[1];
        float gz = R[6]*lq[0] + R[7]*lq[1] + R[8]*lq[2] + t[2];
        atomicAdd(&sqn[h], gx*gx + gy*gy + gz*gz);
        bf16_t* qh = qhat + ((size_t)i * H_ + h) * 32 + 16 + (tid & 3) * 3;
        qh[0] = f2bf(sg[h]*gx); qh[1] = f2bf(sg[h]*gy); qh[2] = f2bf(sg[h]*gz);
        const float* lk = kp + (size_t)i * 144 + tid * 3;
        float hx = R[0]*lk[0] + R[1]*lk[1] + R[2]*lk[2] + t[0];
        float hy = R[3]*lk[0] + R[4]*lk[1] + R[5]*lk[2] + t[1];
        float hz = R[6]*lk[0] + R[7]*lk[1] + R[8]*lk[2] + t[2];
        atomicAdd(&skn[h], hx*hx + hy*hy + hz*hz);
        bf16_t* kh = khat + ((size_t)i * H_ + h) * 32 + 16 + (tid & 3) * 3;
        kh[0] = f2bf(sg[h]*hx); kh[1] = f2bf(sg[h]*hy); kh[2] = f2bf(sg[h]*hz);
        // zero pad dims 28..31
        int c = 28 + (tid & 3);
        qhat[((size_t)i * H_ + h) * 32 + c] = __builtin_bit_cast(bf16_t, (unsigned short)0);
        khat[((size_t)i * H_ + h) * 32 + c] = __builtin_bit_cast(bf16_t, (unsigned short)0);
    }
    if (tid < 96) { // (h,p) for v points -> vB rows 16..39
        int h = tid >> 3, p = tid & 7;
        const float* lv = vp + (size_t)i * 288 + tid * 3;
        float gx = R[0]*lv[0] + R[1]*lv[1] + R[2]*lv[2] + t[0];
        float gy = R[3]*lv[0] + R[4]*lv[1] + R[5]*lv[2] + t[1];
        float gz = R[6]*lv[0] + R[7]*lv[1] + R[8]*lv[2] + t[2];
        int nb = 16 + p * 3;
        vB[((size_t)h * 48 + nb + 0) * N_ + i] = f2bf(gx);
        vB[((size_t)h * 48 + nb + 1) * N_ + i] = f2bf(gy);
        vB[((size_t)h * 48 + nb + 2) * N_ + i] = f2bf(gz);
    }
    { // scalar q/k channels (tid < 192 always)
        int h = tid / 16, c = tid % 16;
        qhat[((size_t)i * H_ + h) * 32 + c] = f2bf(0.25f * q[(size_t)i * 192 + tid]);
        khat[((size_t)i * H_ + h) * 32 + c] = f2bf(k[(size_t)i * 192 + tid]);
        vB[((size_t)h * 48 + c) * N_ + i] = f2bf(v[(size_t)i * 192 + tid]);
    }
    __syncthreads();
    if (tid < H_) {
        qn[(size_t)i * H_ + tid] = sqn[tid];
        kn[(size_t)i * H_ + tid] = skn[tid];
    }
}

// ---------------- bias[i][j][h] = z[i,j,:] @ Wb + bb (WMMA, z cvt in-register) ----------------
__global__ void k_bias(const float* __restrict__ z, const bf16_t* __restrict__ WbTp,
                       const float* __restrict__ bb, bf16_t* __restrict__ bias) {
    int i = blockIdx.x, j0 = blockIdx.y * 16, lane = threadIdx.x;
    v8f acc = {0, 0, 0, 0, 0, 0, 0, 0};
    int mm = lane & 15;
    const float* zrow = z + ((size_t)i * N_ + j0 + mm) * CZ_;
    __builtin_prefetch(z + ((size_t)i * N_ + j0 + 16) * CZ_, 0, 0);
    for (int k0 = 0; k0 < CZ_; k0 += 32) {
        int koff = k0 + ((lane >> 4) << 3);
        const float* p = zrow + koff;
        v16bf af;
        for (int e = 0; e < 8; e++) af[e] = f2bf(p[e]);
        for (int e = 0; e < 8; e++) af[8 + e] = f2bf(p[16 + e]);
        v16bf bfm = load_b_frag(WbTp, CZ_, 0, k0, lane);
        acc = wmma_bf16(af, bfm, acc);
    }
    int h = lane & 15;
    if (h < H_) {
        for (int r = 0; r < 8; r++) {
            int j = j0 + r + ((lane >> 4) << 3);
            bias[((size_t)i * N_ + j) * H_ + h] = f2bf(acc[r] + bb[h]);
        }
    }
}

// ---------------- logits (WMMA) + softmax -> a bf16 [i][h][j] ----------------
__global__ void k_attn(const bf16_t* __restrict__ qhat, const bf16_t* __restrict__ khat,
                       const bf16_t* __restrict__ bias, const float* __restrict__ qn,
                       const float* __restrict__ kn, const float* __restrict__ head_w,
                       bf16_t* __restrict__ aout) {
    __shared__ float L[16 * N_];
    __shared__ float knl[N_];
    __shared__ float qt[16];
    int i0 = blockIdx.x * 16, h = blockIdx.y, lane = threadIdx.x;
    float gam = logf(1.f + expf(head_w[h]));
    float ch = 0.5f * WC_f * gam;
    for (int j = lane; j < N_; j += 32) knl[j] = ch * kn[(size_t)j * H_ + h];
    if (lane < 16) qt[lane] = ch * qn[(size_t)(i0 + lane) * H_ + h];
    __syncthreads();
    const bf16_t* Ae = qhat + (size_t)i0 * (H_ * 32) + h * 32;
    const bf16_t* Be = khat + h * 32;
    v16bf af = load_a_frag(Ae, H_ * 32, 0, 0, lane);
    for (int j0 = 0; j0 < N_; j0 += 16) {
        v16bf bfm = load_b_frag(Be, H_ * 32, j0, 0, lane);
        v8f c = {0, 0, 0, 0, 0, 0, 0, 0};
        c = wmma_bf16(af, bfm, c);
        int n = lane & 15;
        int j = j0 + n;
        for (int r = 0; r < 8; r++) {
            int m = r + ((lane >> 4) << 3);
            float bi = bf2f(bias[((size_t)(i0 + m) * N_ + j) * H_ + h]);
            L[m * N_ + j] = WL_f * (c[r] + bi - qt[m] - knl[j]);
        }
    }
    __syncthreads();
    for (int m = 0; m < 16; m++) {
        float mx = -1e30f;
        for (int j = lane; j < N_; j += 32) mx = fmaxf(mx, L[m * N_ + j]);
        for (int s = 16; s > 0; s >>= 1) mx = fmaxf(mx, __shfl_xor(mx, s));
        float sum = 0.f;
        for (int j = lane; j < N_; j += 32) {
            float e = __expf(L[m * N_ + j] - mx);
            L[m * N_ + j] = e;
            sum += e;
        }
        for (int s = 16; s > 0; s >>= 1) sum += __shfl_xor(sum, s);
        float inv = 1.f / sum;
        bf16_t* dst = aout + ((size_t)(i0 + m) * H_ + h) * N_;
        for (int j = lane; j < N_; j += 32) dst[j] = f2bf(L[m * N_ + j] * inv);
    }
}

// ---------------- o and opg: A=a[i][h][j], B=[v|vpg] packed per head ----------------
__global__ void k_ov(const bf16_t* __restrict__ a, const bf16_t* __restrict__ vB,
                     float* __restrict__ feats, float* __restrict__ opg) {
    int i0 = blockIdx.x * 16, h = blockIdx.y, lane = threadIdx.x;
    const bf16_t* Ae = a + ((size_t)i0 * H_ + h) * N_;
    const bf16_t* Be = vB + (size_t)h * 48 * N_;
    v8f c0 = {0,0,0,0,0,0,0,0}, c1 = {0,0,0,0,0,0,0,0}, c2 = {0,0,0,0,0,0,0,0};
    for (int k0 = 0; k0 < N_; k0 += 32) {
        v16bf af = load_a_frag(Ae, H_ * N_, 0, k0, lane);
        c0 = wmma_bf16(af, load_b_frag(Be, N_, 0,  k0, lane), c0);
        c1 = wmma_bf16(af, load_b_frag(Be, N_, 16, k0, lane), c1);
        c2 = wmma_bf16(af, load_b_frag(Be, N_, 32, k0, lane), c2);
    }
    int n = lane & 15;
    for (int r = 0; r < 8; r++) {
        int m = i0 + r + ((lane >> 4) << 3);
        feats[(size_t)m * FEAT_ + h * 16 + n] = c0[r];
        opg[((size_t)m * H_ + h) * 24 + n] = c1[r];
        if (n < 8) opg[((size_t)m * H_ + h) * 24 + 16 + n] = c2[r];
    }
}

// ---------------- o_pair[i,h,:] = a_h[i,:] @ z[i,:,:]  (M=h, K=j, N=zc) ----------------
// z tile (32 rows x 128 f32, 16KB contiguous) staged into LDS via the Tensor Data
// Mover (TENSOR_LOAD_TO_LDS, tracked by TENSORcnt) when the builtin is available.
__global__ void k_opair(const bf16_t* __restrict__ a, const float* __restrict__ z,
                        float* __restrict__ feats) {
    __shared__ float zt[32 * CZ_];
    int i = blockIdx.x, lane = threadIdx.x;
    v8f zero = {0,0,0,0,0,0,0,0};
    v8f acc[8];
    for (int nt = 0; nt < 8; nt++) acc[nt] = zero;
    int m = lane & 15;
    const bf16_t* Ae = a + (size_t)i * H_ * N_;
    for (int j0 = 0; j0 < N_; j0 += 32) {
        const float* gsrc = z + ((size_t)i * N_ + j0) * CZ_;
#if defined(__has_builtin) && __has_builtin(__builtin_amdgcn_tensor_load_to_lds)
        {
            // D# group0: count=1 | lds_addr | global_addr(57b) | type=2
            unsigned long long ga = (unsigned long long)(uintptr_t)gsrc;
            unsigned ldsoff = (unsigned)(uintptr_t)&zt[0];
            u32x4 g0;
            g0[0] = 1u;                                   // count=1, user mode
            g0[1] = ldsoff;                               // lds_addr
            g0[2] = (unsigned)(ga & 0xFFFFFFFFu);         // global_addr[31:0]
            g0[3] = (unsigned)(((ga >> 32) & 0x01FFFFFFu) // global_addr[56:32]
                               | 0x80000000u);            // type=2 ("image")
            // D# group1: data_size=4B, tensor 128 x 768, tile 128 x 32, stride 128
            i32x8 g1;
            g1[0] = 0x00020000;          // workgroup_mask=0, data_size=2 (4B)
            g1[1] = (int)(128u << 16);   // tensor_dim0[15:0]=128
            g1[2] = (int)(768u << 16);   // tensor_dim0[31:16]=0, tensor_dim1[15:0]=768
            g1[3] = (int)(128u << 16);   // tensor_dim1[31:16]=0, tile_dim0=128
            g1[4] = 32;                  // tile_dim1=32, tile_dim2=0
            g1[5] = 128;                 // tensor_dim0_stride[31:0]=128
            g1[6] = 0;                   // stride hi, tensor_dim1_stride lo
            g1[7] = 0;
            i32x4 gz2 = {0, 0, 0, 0};
            i32x4 gz3 = {0, 0, 0, 0};
#if __clang_major__ >= 23
            i32x8 gz4 = {0, 0, 0, 0, 0, 0, 0, 0};
            __builtin_amdgcn_tensor_load_to_lds(g0, g1, gz2, gz3, gz4, 0);
#else
            __builtin_amdgcn_tensor_load_to_lds(g0, g1, gz2, gz3, 0);
#endif
            __builtin_amdgcn_s_wait_tensorcnt(0);
        }
#else
        {
            const f4* src = (const f4*)gsrc;
            f4* dst4 = (f4*)zt;
            for (int tI = 0; tI < 32; tI++) dst4[lane + 32 * tI] = src[lane + 32 * tI];
        }
#endif
        if (j0 + 32 < N_)
            __builtin_prefetch(z + ((size_t)i * N_ + j0 + 32) * CZ_, 0, 0);
        __syncthreads();
        v16bf af;
        if (m < H_) {
            af = load_a_frag(Ae, N_, 0, j0, lane);
        } else {
            for (int e = 0; e < 16; e++) af[e] = __builtin_bit_cast(bf16_t, (unsigned short)0);
        }
        int kb = ((lane >> 4) << 3);
        for (int nt = 0; nt < 8; nt++) {
            int zc = nt * 16 + (lane & 15);
            v16bf bfm;
            for (int e = 0; e < 16; e++) bfm[e] = f2bf(zt[(kb + e + ((e >> 3) << 3)) * CZ_ + zc]);
            acc[nt] = wmma_bf16(af, bfm, acc[nt]);
        }
        __syncthreads();
    }
    for (int nt = 0; nt < 8; nt++) {
        for (int r = 0; r < 8; r++) {
            int hh = r + ((lane >> 4) << 3);
            if (hh < H_)
                feats[(size_t)i * FEAT_ + 576 + hh * CZ_ + nt * 16 + (lane & 15)] = acc[nt][r];
        }
    }
}

// ---------------- o_pts / o_norm ----------------
__global__ void k_opts(const float* __restrict__ opg, const float* __restrict__ rot,
                       const float* __restrict__ trans, float* __restrict__ feats) {
    int i = blockIdx.x, tid = threadIdx.x; // 96 threads: (h*8+p)
    if (tid >= 96) return;
    int h = tid >> 3;
    const float* R = rot + (size_t)i * 9;
    const float* t = trans + (size_t)i * 3;
    const float* o = opg + ((size_t)i * H_ + h) * 24 + (tid & 7) * 3;
    float dx = o[0] - t[0], dy = o[1] - t[1], dz = o[2] - t[2];
    float px = R[0]*dx + R[3]*dy + R[6]*dz;
    float py = R[1]*dx + R[4]*dy + R[7]*dz;
    float pz = R[2]*dx + R[5]*dy + R[8]*dz;
    float* f = feats + (size_t)i * FEAT_;
    f[192 + tid * 3 + 0] = px;
    f[192 + tid * 3 + 1] = py;
    f[192 + tid * 3 + 2] = pz;
    f[480 + tid] = sqrtf(px*px + py*py + pz*pz + 1e-8f);
}

// ---------------- residual + layernorm (D=384), optional bf16 emit ----------------
__global__ void k_res_ln(const float* __restrict__ x, const float* __restrict__ res,
                         const float* __restrict__ g, const float* __restrict__ b,
                         float* __restrict__ y, bf16_t* __restrict__ ybf) {
    int i = blockIdx.x, tid = threadIdx.x; // 128 threads
    __shared__ float red[128];
    size_t base = (size_t)i * CS_;
    float e0 = x[base + tid] + res[base + tid];
    float e1 = x[base + tid + 128] + res[base + tid + 128];
    float e2 = x[base + tid + 256] + res[base + tid + 256];
    red[tid] = e0 + e1 + e2;
    __syncthreads();
    for (int s = 64; s > 0; s >>= 1) { if (tid < s) red[tid] += red[tid + s]; __syncthreads(); }
    float mean = red[0] / (float)CS_;
    __syncthreads();
    float c0 = e0 - mean, c1 = e1 - mean, c2 = e2 - mean;
    red[tid] = c0*c0 + c1*c1 + c2*c2;
    __syncthreads();
    for (int s = 64; s > 0; s >>= 1) { if (tid < s) red[tid] += red[tid + s]; __syncthreads(); }
    float rstd = rsqrtf(red[0] / (float)CS_ + 1e-5f);
    float v0 = c0 * rstd * g[tid]       + b[tid];
    float v1 = c1 * rstd * g[tid + 128] + b[tid + 128];
    float v2 = c2 * rstd * g[tid + 256] + b[tid + 256];
    y[base + tid]       = v0;
    y[base + tid + 128] = v1;
    y[base + tid + 256] = v2;
    if (ybf) {
        ybf[base + tid]       = f2bf(v0);
        ybf[base + tid + 128] = f2bf(v1);
        ybf[base + tid + 256] = f2bf(v2);
    }
}

// ---------------- frame update: u -> quat -> rot_n, trans_n ----------------
__global__ void k_frame(const float* __restrict__ sfin, const float* __restrict__ Wu,
                        const float* __restrict__ bu, const float* __restrict__ rot,
                        const float* __restrict__ trans, float* __restrict__ outR,
                        float* __restrict__ outT, float* __restrict__ lossSlot) {
    int i = blockIdx.x, lane = threadIdx.x;
    float u[6] = {0, 0, 0, 0, 0, 0};
    for (int k = lane; k < CS_; k += 32) {
        float sv = sfin[(size_t)i * CS_ + k];
        for (int o = 0; o < 6; o++) u[o] += sv * Wu[k * 6 + o];
    }
    for (int s = 16; s > 0; s >>= 1)
        for (int o = 0; o < 6; o++) u[o] += __shfl_xor(u[o], s);
    if (lane == 0) {
        for (int o = 0; o < 6; o++) u[o] += bu[o];
        float a = 1.f, b = u[0], c = u[1], d = u[2];
        float inv = rsqrtf(a*a + b*b + c*c + d*d);
        a *= inv; b *= inv; c *= inv; d *= inv;
        float ru[9] = { a*a + b*b - c*c - d*d, 2.f*(b*c - a*d),       2.f*(b*d + a*c),
                        2.f*(b*c + a*d),       a*a - b*b + c*c - d*d, 2.f*(c*d - a*b),
                        2.f*(b*d - a*c),       2.f*(c*d + a*b),       a*a - b*b - c*c + d*d };
        const float* R = rot + (size_t)i * 9;
        float* Rn = outR + (size_t)i * 9;
        for (int r = 0; r < 3; r++)
            for (int kk = 0; kk < 3; kk++)
                Rn[r * 3 + kk] = R[r*3+0]*ru[0*3+kk] + R[r*3+1]*ru[1*3+kk] + R[r*3+2]*ru[2*3+kk];
        const float* t = trans + (size_t)i * 3;
        float* Tn = outT + (size_t)i * 3;
        for (int r = 0; r < 3; r++)
            Tn[r] = R[r*3+0]*u[3] + R[r*3+1]*u[4] + R[r*3+2]*u[5] + t[r];
        if (i == 0) *lossSlot = 0.f;
    }
}

// ---------------- clamped FAPE loss ----------------
__global__ void k_fape(const float* __restrict__ Rn, const float* __restrict__ Tn,
                       const float* __restrict__ Rt, const float* __restrict__ Tt,
                       const int* __restrict__ clampP, float* __restrict__ lossSlot) {
    int i = blockIdx.x, tid = threadIdx.x; // 256 threads
    __shared__ float red[256];
    float fc = (float)(*clampP);
    const float* Ri = Rn + (size_t)i * 9;
    const float* Gi = Rt + (size_t)i * 9;
    float tix = Tn[i*3+0], tiy = Tn[i*3+1], tiz = Tn[i*3+2];
    float gix = Tt[i*3+0], giy = Tt[i*3+1], giz = Tt[i*3+2];
    float acc = 0.f;
    for (int j = tid; j < N_; j += 256) {
        float dx = Tn[j*3+0] - tix, dy = Tn[j*3+1] - tiy, dz = Tn[j*3+2] - tiz;
        float px = Ri[0]*dx + Ri[3]*dy + Ri[6]*dz;
        float py = Ri[1]*dx + Ri[4]*dy + Ri[7]*dz;
        float pz = Ri[2]*dx + Ri[5]*dy + Ri[8]*dz;
        float ex = Tt[j*3+0] - gix, ey = Tt[j*3+1] - giy, ez = Tt[j*3+2] - giz;
        float qx = Gi[0]*ex + Gi[3]*ey + Gi[6]*ez;
        float qy = Gi[1]*ex + Gi[4]*ey + Gi[7]*ez;
        float qz = Gi[2]*ex + Gi[5]*ey + Gi[8]*ez;
        float ux = px - qx, uy = py - qy, uz = pz - qz;
        float dd = sqrtf(ux*ux + uy*uy + uz*uz + 1e-12f);
        acc += fminf(dd, fc);
    }
    red[tid] = acc;
    __syncthreads();
    for (int s = 128; s > 0; s >>= 1) { if (tid < s) red[tid] += red[tid + s]; __syncthreads(); }
    if (tid == 0) atomicAdd(lossSlot, red[0] * (1.f / ((float)N_ * (float)N_ * 10.f)));
}

// ==================================================================================
extern "C" void kernel_launch(void* const* d_in, const int* in_sizes, int n_in,
                              void* d_out, int out_size, void* d_ws, size_t ws_size,
                              hipStream_t stream) {
    (void)in_sizes; (void)n_in; (void)out_size; (void)ws_size;
    const float* single  = (const float*)d_in[0];
    const float* pair    = (const float*)d_in[1];
    const float* rot     = (const float*)d_in[2];
    const float* trans   = (const float*)d_in[3];
    const float* rot_t   = (const float*)d_in[4];
    const float* trans_t = (const float*)d_in[5];
    const float* Wq  = (const float*)d_in[6];  const float* bq  = (const float*)d_in[7];
    const float* Wk  = (const float*)d_in[8];  const float* bk  = (const float*)d_in[9];
    const float* Wv  = (const float*)d_in[10]; const float* bv  = (const float*)d_in[11];
    const float* Wqp = (const float*)d_in[12]; const float* bqp = (const float*)d_in[13];
    const float* Wkp = (const float*)d_in[14]; const float* bkp = (const float*)d_in[15];
    const float* Wvp = (const float*)d_in[16]; const float* bvp = (const float*)d_in[17];
    const float* Wb  = (const float*)d_in[18]; const float* bb  = (const float*)d_in[19];
    const float* Wo  = (const float*)d_in[20]; const float* bo  = (const float*)d_in[21];
    const float* head_w = (const float*)d_in[22];
    const float* Wt1 = (const float*)d_in[23]; const float* bt1 = (const float*)d_in[24];
    const float* Wt2 = (const float*)d_in[25]; const float* bt2 = (const float*)d_in[26];
    const float* Wt3 = (const float*)d_in[27]; const float* bt3 = (const float*)d_in[28];
    const float* Wu  = (const float*)d_in[29]; const float* bu  = (const float*)d_in[30];
    const float* g1  = (const float*)d_in[31]; const float* b1  = (const float*)d_in[32];
    const float* g2  = (const float*)d_in[33]; const float* b2  = (const float*)d_in[34];
    const int* fclamp = (const int*)d_in[35];

    char* wsp = (char*)d_ws;
    size_t off = 0;
    auto alloc = [&](size_t bytes) -> void* {
        void* p = wsp + off;
        off += (bytes + 255) & ~(size_t)255;
        return p;
    };
    bf16_t* s_bf   = (bf16_t*)alloc((size_t)N_ * CS_ * 2);
    bf16_t* WqT    = (bf16_t*)alloc((size_t)192 * 384 * 2);
    bf16_t* WkT    = (bf16_t*)alloc((size_t)192 * 384 * 2);
    bf16_t* WvT    = (bf16_t*)alloc((size_t)192 * 384 * 2);
    bf16_t* WqpT   = (bf16_t*)alloc((size_t)144 * 384 * 2);
    bf16_t* WkpT   = (bf16_t*)alloc((size_t)144 * 384 * 2);
    bf16_t* WvpT   = (bf16_t*)alloc((size_t)288 * 384 * 2);
    bf16_t* WbTp   = (bf16_t*)alloc((size_t)16 * 128 * 2);
    bf16_t* WoT    = (bf16_t*)alloc((size_t)384 * FEAT_ * 2);
    bf16_t* Wt1T   = (bf16_t*)alloc((size_t)384 * 384 * 2);
    bf16_t* Wt2T   = (bf16_t*)alloc((size_t)384 * 384 * 2);
    bf16_t* Wt3T   = (bf16_t*)alloc((size_t)384 * 384 * 2);
    float*  qbuf   = (float*)alloc((size_t)N_ * 192 * 4);
    float*  kbuf   = (float*)alloc((size_t)N_ * 192 * 4);
    float*  vbuf   = (float*)alloc((size_t)N_ * 192 * 4);
    float*  qpbuf  = (float*)alloc((size_t)N_ * 144 * 4);
    float*  kpbuf  = (float*)alloc((size_t)N_ * 144 * 4);
    float*  vpbuf  = (float*)alloc((size_t)N_ * 288 * 4);
    bf16_t* qhat   = (bf16_t*)alloc((size_t)N_ * H_ * 32 * 2);
    bf16_t* khat   = (bf16_t*)alloc((size_t)N_ * H_ * 32 * 2);
    float*  qn     = (float*)alloc((size_t)N_ * H_ * 4);
    float*  kn     = (float*)alloc((size_t)N_ * H_ * 4);
    bf16_t* vB     = (bf16_t*)alloc((size_t)H_ * 48 * N_ * 2);
    bf16_t* biasbf = (bf16_t*)alloc((size_t)N_ * N_ * H_ * 2);
    bf16_t* abf    = (bf16_t*)alloc((size_t)N_ * H_ * N_ * 2);
    float*  opg    = (float*)alloc((size_t)N_ * H_ * 24 * 4);
    float*  feats  = (float*)alloc((size_t)N_ * FEAT_ * 4);
    bf16_t* featsb = (bf16_t*)alloc((size_t)N_ * FEAT_ * 2);
    float*  oipa   = (float*)alloc((size_t)N_ * CS_ * 4);
    float*  sln    = (float*)alloc((size_t)N_ * CS_ * 4);
    bf16_t* slnb   = (bf16_t*)alloc((size_t)N_ * CS_ * 2);
    float*  h1     = (float*)alloc((size_t)N_ * CS_ * 4);
    bf16_t* h1b    = (bf16_t*)alloc((size_t)N_ * CS_ * 2);
    float*  h2     = (float*)alloc((size_t)N_ * CS_ * 4);
    bf16_t* h2b    = (bf16_t*)alloc((size_t)N_ * CS_ * 2);
    float*  h3     = (float*)alloc((size_t)N_ * CS_ * 4);

    float* outF = (float*)d_out;
    const int OUT_R = N_ * CS_;          // 294912
    const int OUT_T = OUT_R + N_ * 9;    // 301824
    const int OUT_L = OUT_T + N_ * 3;    // 304128

    // --- precision prep: bf16 activations + transposed bf16 weights ---
    int ns = N_ * CS_;
    k_cvt_bf16<<<(ns + 255) / 256, 256, 0, stream>>>(single, s_bf, ns);
    k_transpose<<<(384 * 192 + 255) / 256, 256, 0, stream>>>(Wq, WqT, 384, 192);
    k_transpose<<<(384 * 192 + 255) / 256, 256, 0, stream>>>(Wk, WkT, 384, 192);
    k_transpose<<<(384 * 192 + 255) / 256, 256, 0, stream>>>(Wv, WvT, 384, 192);
    k_transpose<<<(384 * 144 + 255) / 256, 256, 0, stream>>>(Wqp, WqpT, 384, 144);
    k_transpose<<<(384 * 144 + 255) / 256, 256, 0, stream>>>(Wkp, WkpT, 384, 144);
    k_transpose<<<(384 * 288 + 255) / 256, 256, 0, stream>>>(Wvp, WvpT, 384, 288);
    k_zero_bf16<<<(16 * 128 + 255) / 256, 256, 0, stream>>>(WbTp, 16 * 128);
    k_transpose<<<(128 * 12 + 255) / 256, 256, 0, stream>>>(Wb, WbTp, 128, 12);
    k_transpose<<<(FEAT_ * 384 + 1023) / 1024, 1024, 0, stream>>>(Wo, WoT, FEAT_, 384);
    k_transpose<<<(384 * 384 + 255) / 256, 256, 0, stream>>>(Wt1, Wt1T, 384, 384);
    k_transpose<<<(384 * 384 + 255) / 256, 256, 0, stream>>>(Wt2, Wt2T, 384, 384);
    k_transpose<<<(384 * 384 + 255) / 256, 256, 0, stream>>>(Wt3, Wt3T, 384, 384);
    k_zero_bf16<<<((int)(H_ * 48 * N_) + 255) / 256, 256, 0, stream>>>(vB, H_ * 48 * N_);

    // --- projections (WMMA GEMMs) ---
    k_gemm<<<dim3(48, 12), 32, 0, stream>>>(s_bf, WqT, bq, qbuf, nullptr, N_, 384, 192, 0);
    k_gemm<<<dim3(48, 12), 32, 0, stream>>>(s_bf, WkT, bk, kbuf, nullptr, N_, 384, 192, 0);
    k_gemm<<<dim3(48, 12), 32, 0, stream>>>(s_bf, WvT, bv, vbuf, nullptr, N_, 384, 192, 0);
    k_gemm<<<dim3(48,  9), 32, 0, stream>>>(s_bf, WqpT, bqp, qpbuf, nullptr, N_, 384, 144, 0);
    k_gemm<<<dim3(48,  9), 32, 0, stream>>>(s_bf, WkpT, bkp, kpbuf, nullptr, N_, 384, 144, 0);
    k_gemm<<<dim3(48, 18), 32, 0, stream>>>(s_bf, WvpT, bvp, vpbuf, nullptr, N_, 384, 288, 0);

    // --- IPA ---
    k_prep<<<N_, 192, 0, stream>>>(qbuf, kbuf, vbuf, qpbuf, kpbuf, vpbuf, rot, trans,
                                   head_w, qhat, khat, vB, qn, kn);
    k_bias<<<dim3(N_, 48), 32, 0, stream>>>(pair, WbTp, bb, biasbf);
    k_attn<<<dim3(48, 12), 32, 0, stream>>>(qhat, khat, biasbf, qn, kn, head_w, abf);
    k_ov<<<dim3(48, 12), 32, 0, stream>>>(abf, vB, feats, opg);
    k_opair<<<N_, 32, 0, stream>>>(abf, pair, feats);
    k_opts<<<N_, 96, 0, stream>>>(opg, rot, trans, feats);

    // --- output projection + residual/LN + transition + LN ---
    int nf = N_ * FEAT_;
    k_cvt_bf16<<<(nf + 255) / 256, 256, 0, stream>>>(feats, featsb, nf);
    k_gemm<<<dim3(48, 24), 32, 0, stream>>>(featsb, WoT, bo, oipa, nullptr, N_, FEAT_, 384, 0);
    k_res_ln<<<N_, 128, 0, stream>>>(oipa, single, g1, b1, sln, slnb);
    k_gemm<<<dim3(48, 24), 32, 0, stream>>>(slnb, Wt1T, bt1, h1, h1b, N_, 384, 384, 1);
    k_gemm<<<dim3(48, 24), 32, 0, stream>>>(h1b, Wt2T, bt2, h2, h2b, N_, 384, 384, 1);
    k_gemm<<<dim3(48, 24), 32, 0, stream>>>(h2b, Wt3T, bt3, h3, nullptr, N_, 384, 384, 0);
    k_res_ln<<<N_, 128, 0, stream>>>(h3, sln, g2, b2, outF, nullptr);

    // --- frame update + FAPE loss ---
    k_frame<<<N_, 32, 0, stream>>>(outF, Wu, bu, rot, trans,
                                   outF + OUT_R, outF + OUT_T, outF + OUT_L);
    k_fape<<<N_, 256, 0, stream>>>(outF + OUT_R, outF + OUT_T, rot_t, trans_t,
                                   fclamp, outF + OUT_L);
}
